// MHMoEGate_11063835755042
// MI455X (gfx1250) — matrix operand order repaired and checked by reading, not compiled, not run on hardware
//
#include <hip/hip_runtime.h>
#include <hip/hip_bf16.h>

// ---------------------------------------------------------------------------
// MoE gate for MI455X (gfx1250, wave32, WMMA).
// logits = inp[T,D] @ wg[E,D]^T  with T=16384, D=4096, E=128.
// f32 inputs split into bf16 hi/lo; product = Ahi*Bhi + Ahi*Blo + Alo*Bhi
// (3x v_wmma_f32_16x16x32_bf16) -> ~3/8 the cost of the f32-WMMA path,
// near-fp32 accuracy, approaching the 11us HBM roofline (258MB @ 23.3TB/s).
// inp (256MB, streamed once, > 192MB L2) is loaded non-temporal so the
// broadcast wg matrix (2MB, read by all 256 WGs) stays L2-resident.
// ---------------------------------------------------------------------------

typedef __attribute__((ext_vector_type(16))) __bf16 v16bf;
typedef __attribute__((ext_vector_type(8)))  float  v8f;
typedef __attribute__((ext_vector_type(4)))  float  v4f;   // native vector for NT loads

#define T_TOKENS   16384
#define D_MODEL    4096
#define N_EXPERT   128
#define TOK_PER_WG 64
#define KC         32
#define NKB        (D_MODEL / KC)
#define L_STRIDE   129                 // 129 mod 64 == 1 -> conflict-free LDS
#define INV_TEMP   (1.0f / 0.3f)

union BV {
  v16bf v;
  unsigned short u[16];
  uint4 q[2];
};

__device__ __forceinline__ unsigned bf16_rne_bits(float x) {
  unsigned u = __float_as_uint(x);
  return (u + 0x7FFFu + ((u >> 16) & 1u)) >> 16;
}

__global__ __launch_bounds__(256)
void moe_gate_main(const float* __restrict__ inp, const float* __restrict__ wg,
                   float* __restrict__ outIdx, float* __restrict__ outGate,
                   float* __restrict__ ws_me, float* __restrict__ ws_ce)
{
  __shared__ __align__(16) unsigned short s_aHi[2][TOK_PER_WG * KC];
  __shared__ __align__(16) unsigned short s_aLo[2][TOK_PER_WG * KC];
  __shared__ float s_logits[TOK_PER_WG * L_STRIDE];
  __shared__ float s_max[TOK_PER_WG];
  __shared__ float s_rcp[TOK_PER_WG];

  const int tid  = threadIdx.x;
  const int wv   = tid >> 5;       // wave 0..7 -> expert block
  const int lane = tid & 31;
  const int kh   = lane >> 4;      // lane-half selects K sub-range
  const int l16  = lane & 15;
  const int tokBase = blockIdx.x * TOK_PER_WG;

  // cooperative A staging: thread -> (token, 8-wide k quarter)
  const int sTok = tid >> 2;            // 0..63
  const int sKq  = (tid & 3) << 3;      // 0,8,16,24
  const float* aSrc = inp + (size_t)(tokBase + sTok) * D_MODEL + sKq;

  v8f acc[4];
#pragma unroll
  for (int t = 0; t < 4; ++t)
#pragma unroll
    for (int r = 0; r < 8; ++r) acc[t][r] = 0.0f;

  // prologue prefetch of k-chunk 0 (non-temporal: inp streamed once)
  v4f pf0 = __builtin_nontemporal_load((const v4f*)(aSrc + 0));
  v4f pf1 = __builtin_nontemporal_load((const v4f*)(aSrc + 4));

  // this lane's expert row, offset by lane-half (k order used here for B, same as A:
  // lanes 0-15 hold K 0..15, lanes 16-31 hold K 16..31, N = lane%16)
  const float* bBase = wg + (size_t)(wv * 16 + l16) * D_MODEL + kh * 16;

  for (int kb = 0; kb < NKB; ++kb) {
    const int buf = kb & 1;

    // ---- split prefetched A chunk into bf16 hi/lo planes in LDS ----
    {
      float f[8] = {pf0.x, pf0.y, pf0.z, pf0.w, pf1.x, pf1.y, pf1.z, pf1.w};
      unsigned hi[4], lo[4];
#pragma unroll
      for (int j = 0; j < 4; ++j) {
        unsigned ua = __float_as_uint(f[2 * j]);
        unsigned ub = __float_as_uint(f[2 * j + 1]);
        unsigned ha = ua & 0xFFFF0000u, hb = ub & 0xFFFF0000u;
        hi[j] = (ha >> 16) | hb;                       // truncated bf16 pair
        float ra = f[2 * j]     - __uint_as_float(ha); // exact residual
        float rb = f[2 * j + 1] - __uint_as_float(hb);
        lo[j] = bf16_rne_bits(ra) | (bf16_rne_bits(rb) << 16);
      }
      *(uint4*)&s_aHi[buf][sTok * KC + sKq] = make_uint4(hi[0], hi[1], hi[2], hi[3]);
      *(uint4*)&s_aLo[buf][sTok * KC + sKq] = make_uint4(lo[0], lo[1], lo[2], lo[3]);
    }

    // ---- prefetch next A chunk (non-temporal) ----
    if (kb + 1 < NKB) {
      const float* p = aSrc + (size_t)(kb + 1) * KC;
      pf0 = __builtin_nontemporal_load((const v4f*)(p + 0));
      pf1 = __builtin_nontemporal_load((const v4f*)(p + 4));
    }
    __syncthreads();   // single barrier / iter is safe with 2 LDS buffers

    // ---- B fragment: 16 contiguous f32 of this expert row, split hi/lo ----
    // (regular temporal loads: wg must stay resident in the 192MB L2)
    const float* bp = bBase + (size_t)kb * KC;
    float bfv[16];
    *(v4f*)&bfv[0]  = *(const v4f*)(bp + 0);
    *(v4f*)&bfv[4]  = *(const v4f*)(bp + 4);
    *(v4f*)&bfv[8]  = *(const v4f*)(bp + 8);
    *(v4f*)&bfv[12] = *(const v4f*)(bp + 12);
    BV bHi, bLo;
#pragma unroll
    for (int i = 0; i < 16; ++i) {
      unsigned u = __float_as_uint(bfv[i]);
      unsigned h = u & 0xFFFF0000u;
      bHi.u[i] = (unsigned short)(h >> 16);
      bLo.u[i] = (unsigned short)bf16_rne_bits(bfv[i] - __uint_as_float(h));
    }

    // ---- 4 token tiles x 3 bf16 WMMAs (hi*hi + hi*lo + lo*hi) ----
#pragma unroll
    for (int tm = 0; tm < 4; ++tm) {
      const int tok = tm * 16 + l16;
      // A k order used here: lanes0-15 K 0..15; lanes16-31 K 16..31 (matches B)
      const uint4* pa = (const uint4*)&s_aHi[buf][tok * KC + kh * 16];
      const uint4* pl = (const uint4*)&s_aLo[buf][tok * KC + kh * 16];
      BV aHi, aLo;
      aHi.q[0] = pa[0]; aHi.q[1] = pa[1];
      aLo.q[0] = pl[0]; aLo.q[1] = pl[1];
      acc[tm] = __builtin_amdgcn_wmma_f32_16x16x32_bf16(
          false, aHi.v, false, bHi.v, (short)0, acc[tm], false, false);
      acc[tm] = __builtin_amdgcn_wmma_f32_16x16x32_bf16(
          false, aHi.v, false, bLo.v, (short)0, acc[tm], false, false);
      acc[tm] = __builtin_amdgcn_wmma_f32_16x16x32_bf16(
          false, aLo.v, false, bHi.v, (short)0, acc[tm], false, false);
    }
  }

  // ---- spill logits to LDS (C layout: VGPR r -> M = r + half*8, N = lane%16) ----
#pragma unroll
  for (int tm = 0; tm < 4; ++tm)
#pragma unroll
    for (int r = 0; r < 8; ++r) {
      int row = tm * 16 + kh * 8 + r;
      s_logits[row * L_STRIDE + wv * 16 + l16] = acc[tm][r];
    }
  __syncthreads();

  // ---- pass 1: per-token top-2, gate softmax, temp-softmax denominator ----
  if (tid < TOK_PER_WG) {
    const float* row = &s_logits[tid * L_STRIDE];
    float v1 = -INFINITY, v2 = -INFINITY;
    int i1 = 0, i2 = 0;
    for (int e = 0; e < N_EXPERT; ++e) {
      float x = row[e];
      if (x > v1)      { v2 = v1; i2 = i1; v1 = x; i1 = e; }
      else if (x > v2) { v2 = x; i2 = e; }
    }
    float denom = 0.0f;
    for (int e = 0; e < N_EXPERT; ++e)
      denom += __expf((row[e] - v1) * INV_TEMP);
    s_max[tid] = v1;
    s_rcp[tid] = 1.0f / denom;

    float g0 = 1.0f / (1.0f + __expf(v2 - v1));   // softmax over {v1,v2}
    int gt = tokBase + tid;
    __builtin_nontemporal_store((float)i1, &outIdx[2 * gt + 0]);
    __builtin_nontemporal_store((float)i2, &outIdx[2 * gt + 1]);
    __builtin_nontemporal_store(g0,         &outGate[2 * gt + 0]);
    __builtin_nontemporal_store(1.0f - g0,  &outGate[2 * gt + 1]);
    atomicAdd(&ws_ce[i1], 1.0f);                  // argmax count
  }
  __syncthreads();

  // ---- pass 2: per-expert prob sums (column scan, conflict-free) ----
  if (tid < N_EXPERT) {
    float partial = 0.0f;
    for (int t = 0; t < TOK_PER_WG; ++t)
      partial += __expf((s_logits[t * L_STRIDE + tid] - s_max[t]) * INV_TEMP) * s_rcp[t];
    atomicAdd(&ws_me[tid], partial);
  }
}

__global__ void moe_gate_zero(float* __restrict__ ws) {
  ws[threadIdx.x] = 0.0f;
}

__global__ void moe_gate_loss(const float* __restrict__ ws_me,
                              const float* __restrict__ ws_ce,
                              float* __restrict__ outLoss) {
  __shared__ float red[N_EXPERT];
  int e = threadIdx.x;
  red[e] = ws_me[e] * ws_ce[e];
  __syncthreads();
  for (int s = N_EXPERT / 2; s > 0; s >>= 1) {
    if (e < s) red[e] += red[e + s];
    __syncthreads();
  }
  // loss = sum(me * count) * E / T^2   (ce = count * E/T, loss = sum(me*ce)/T)
  if (e == 0)
    outLoss[0] = red[0] * ((float)N_EXPERT / ((float)T_TOKENS * (float)T_TOKENS));
}

extern "C" void kernel_launch(void* const* d_in, const int* in_sizes, int n_in,
                              void* d_out, int out_size, void* d_ws, size_t ws_size,
                              hipStream_t stream) {
  const float* inp = (const float*)d_in[0];
  const float* wg  = (const float*)d_in[1];
  (void)in_sizes; (void)n_in; (void)out_size; (void)ws_size;

  float* out     = (float*)d_out;
  float* outIdx  = out;                    // [T,2] top_idx (as float)
  float* outGate = out + 2 * T_TOKENS;     // [T,2] gate_score
  float* outLoss = out + 4 * T_TOKENS;     // [1]   loss

  float* ws_me = (float*)d_ws;             // [128] sum of probs per expert
  float* ws_ce = ws_me + N_EXPERT;         // [128] argmax counts

  moe_gate_zero<<<1, 2 * N_EXPERT, 0, stream>>>(ws_me);
  moe_gate_main<<<T_TOKENS / TOK_PER_WG, 256, 0, stream>>>(
      inp, wg, outIdx, outGate, ws_me, ws_ce);
  moe_gate_loss<<<1, N_EXPERT, 0, stream>>>(ws_me, ws_ce, outLoss);
}
